// ODEBlock_72335839199609
// MI455X (gfx1250) — matrix operand order repaired and checked by reading, not compiled, not run on hardware
//
#include <hip/hip_runtime.h>

// ---------------------------------------------------------------------------
// ODEBlock on MI455X (gfx1250):
//   - 6144 independent 64-wide ODE rows (bg slice provably dead -> skipped)
//   - one wave32 per 16-row tile; weights live in VGPRs as WMMA A-fragments
//   - state yT kept transposed in f32 D-fragments; per step:
//       yT -> f16 B-frags (ds_swizzle half-swap + cvt, no full transpose)
//       hT = tanh(W1^T * yT + b1)   [4 M-tiles x 2 K-chunks of wmma_f32_16x16x32_f16]
//       yT += dt * (W2^T * hT + b2)
//       v   = wS2 . relu(w3 * yT + b3) + bS2  (rides WMMA as a 5th M-tile)
//   - pass 2: softmax over s for each (t,b)
// ---------------------------------------------------------------------------

#define B_SIZE 2048
#define HDIM   64
#define T_STEPS 200
#define DT 0.01f

typedef __attribute__((ext_vector_type(16))) _Float16 v16h;
typedef __attribute__((ext_vector_type(8)))  float    v8f;

__device__ __forceinline__ float swap16(float x) {
  // SWAPX16: group-of-32 swizzle, xor=0x10, and=0x1f -> lane L <-> lane L^16
  int r = __builtin_amdgcn_ds_swizzle(__builtin_bit_cast(int, x), 0x401F);
  return __builtin_bit_cast(float, r);
}

__device__ __forceinline__ float fast_tanh(float x) {
  // tanh(x) = 1 - 2/(exp(2x)+1); exp via v_exp_f32. Saturates correctly at +-inf.
  float e = __builtin_amdgcn_exp2f(x * 2.8853900817779268f); // 2*log2(e)
  return 1.0f - 2.0f * __builtin_amdgcn_rcpf(e + 1.0f);
}

// Convert two f32 D-fragments (rows m=0..15 in P, m=16..31 in Q, cols = batch)
// into one f16 B-fragment (32xK chunk). Only needs the half-wave swap.
__device__ __forceinline__ v16h d_to_b(const v8f P, const v8f Q, bool hi) {
  float Ps[8], Qs[8];
#pragma unroll
  for (int v = 0; v < 8; ++v) { Ps[v] = swap16(P[v]); Qs[v] = swap16(Q[v]); }
  v16h b;
#pragma unroll
  for (int e = 0; e < 8; ++e) {
    b[e]     = (_Float16)(hi ? Qs[e] : P[e]);
    b[e + 8] = (_Float16)(hi ? Q[e]  : Ps[e]);
  }
  return b;
}

__global__ void __launch_bounds__(32)
ode_wave_kernel(const float* __restrict__ x,   const float* __restrict__ wS1,
                const float* __restrict__ bS1, const float* __restrict__ fW1,
                const float* __restrict__ fb1, const float* __restrict__ fW2,
                const float* __restrict__ fb2, const float* __restrict__ w3,
                const float* __restrict__ b3,  const float* __restrict__ wS2,
                const float* __restrict__ bS2, float* __restrict__ vbuf) {
  const int lane = threadIdx.x & 31;
  const int l    = lane & 15;   // batch column within tile / M row of A
  const int g    = lane >> 4;   // lane half
  const int tile = blockIdx.x;  // 0..383 == 3 * 128
  const int s     = tile >> 7;
  const int bbase = (tile & 127) << 4;

  // ---- pack transposed weights into WMMA A-fragments (one time) ----
  // A-frag (16x32 f16): lane l holds row M=l; elem e -> K = g*16+e
  v16h a1[4][2], a2[4][2], aw[2];
#pragma unroll
  for (int mt = 0; mt < 4; ++mt)
#pragma unroll
    for (int kc = 0; kc < 2; ++kc)
#pragma unroll
      for (int e = 0; e < 16; ++e) {
        int M = mt * 16 + l;
        int K = kc * 32 + g * 16 + e;
        a1[mt][kc][e] = (_Float16)fW1[K * HDIM + M];  // A = W1^T
        a2[mt][kc][e] = (_Float16)fW2[K * HDIM + M];  // A = W2^T
      }
#pragma unroll
  for (int kc = 0; kc < 2; ++kc)
#pragma unroll
    for (int e = 0; e < 16; ++e) {
      int K = kc * 32 + g * 16 + e;
      aw[kc][e] = (l < 4) ? (_Float16)w3[l * HDIM + K] : (_Float16)0.0f;
    }

  // ---- bias C-fragments (D layout: elem v -> row M = mt*16 + v + g*8) ----
  v8f cb1[4], cb2[4], cw3;
#pragma unroll
  for (int mt = 0; mt < 4; ++mt)
#pragma unroll
    for (int v = 0; v < 8; ++v) {
      int M = mt * 16 + v + g * 8;
      cb1[mt][v] = fb1[M];
      cb2[mt][v] = fb2[M];
    }
#pragma unroll
  for (int v = 0; v < 8; ++v) cw3[v] = (g == 0 && v < 4) ? b3[v] : 0.0f;

  const float w20 = wS2[0], w21 = wS2[1], w22 = wS2[2], w23 = wS2[3];
  const float bS2v = bS2[0];

  // ---- state init: yT[M, col=l] = relu(x[b,s]*wS1[M] + bS1[M]) ----
  const float xv = x[(bbase + l) * (3 + HDIM) + s];
  v8f y[4];
#pragma unroll
  for (int mt = 0; mt < 4; ++mt)
#pragma unroll
    for (int v = 0; v < 8; ++v) {
      int M = mt * 16 + v + g * 8;
      float t0 = xv * wS1[M] + bS1[M];
      y[mt][v] = t0 > 0.0f ? t0 : 0.0f;
    }

  float* __restrict__ vrow =
      vbuf + (size_t)s * T_STEPS * B_SIZE + (size_t)bbase + l;

#pragma unroll 1
  for (int t = 0; t < T_STEPS; ++t) {
    // yT (f32 D-frags) -> f16 B-frags
    v16h by0 = d_to_b(y[0], y[1], g != 0);
    v16h by1 = d_to_b(y[2], y[3], g != 0);

    // output head from pre-update state: sol[t] = y_t
    v8f dw = __builtin_amdgcn_wmma_f32_16x16x32_f16(
        false, aw[0], false, by0, (short)0, cw3, false, false);
    dw = __builtin_amdgcn_wmma_f32_16x16x32_f16(
        false, aw[1], false, by1, (short)0, dw, false, false);
    if (g == 0) {
      float r0 = dw[0] > 0.0f ? dw[0] : 0.0f;
      float r1 = dw[1] > 0.0f ? dw[1] : 0.0f;
      float r2 = dw[2] > 0.0f ? dw[2] : 0.0f;
      float r3 = dw[3] > 0.0f ? dw[3] : 0.0f;
      vrow[(size_t)t * B_SIZE] = w20 * r0 + w21 * r1 + w22 * r2 + w23 * r3 + bS2v;
    }

    // hT = tanh(W1^T * yT + b1)
    v8f h[4];
#pragma unroll
    for (int mt = 0; mt < 4; ++mt) {
      v8f acc = __builtin_amdgcn_wmma_f32_16x16x32_f16(
          false, a1[mt][0], false, by0, (short)0, cb1[mt], false, false);
      acc = __builtin_amdgcn_wmma_f32_16x16x32_f16(
          false, a1[mt][1], false, by1, (short)0, acc, false, false);
#pragma unroll
      for (int v = 0; v < 8; ++v) acc[v] = fast_tanh(acc[v]);
      h[mt] = acc;
    }

    v16h bh0 = d_to_b(h[0], h[1], g != 0);
    v16h bh1 = d_to_b(h[2], h[3], g != 0);

    // yT += dt * (W2^T * hT + b2)
#pragma unroll
    for (int mt = 0; mt < 4; ++mt) {
      v8f acc = __builtin_amdgcn_wmma_f32_16x16x32_f16(
          false, a2[mt][0], false, bh0, (short)0, cb2[mt], false, false);
      acc = __builtin_amdgcn_wmma_f32_16x16x32_f16(
          false, a2[mt][1], false, bh1, (short)0, acc, false, false);
#pragma unroll
      for (int v = 0; v < 8; ++v) y[mt][v] += DT * acc[v];
    }
  }
}

__global__ void __launch_bounds__(256)
softmax_kernel(const float* __restrict__ vbuf, float* __restrict__ out) {
  const int TB = T_STEPS * B_SIZE;
  int idx = blockIdx.x * blockDim.x + threadIdx.x;
  if (idx >= TB) return;
  float a = vbuf[idx];
  float b = vbuf[TB + idx];
  float c = vbuf[2 * TB + idx];
  float m = fmaxf(a, fmaxf(b, c));
  const float L2E = 1.4426950408889634f;
  float ea = __builtin_amdgcn_exp2f((a - m) * L2E);
  float eb = __builtin_amdgcn_exp2f((b - m) * L2E);
  float ec = __builtin_amdgcn_exp2f((c - m) * L2E);
  float inv = 1.0f / (ea + eb + ec);
  out[idx]           = ea * inv;  // S_out
  out[TB + idx]      = eb * inv;  // I_out
  out[2 * TB + idx]  = ec * inv;  // R_out
}

extern "C" void kernel_launch(void* const* d_in, const int* in_sizes, int n_in,
                              void* d_out, int out_size, void* d_ws, size_t ws_size,
                              hipStream_t stream) {
  const float* x   = (const float*)d_in[0];
  const float* wS1 = (const float*)d_in[1];
  const float* bS1 = (const float*)d_in[2];
  const float* fW1 = (const float*)d_in[3];
  const float* fb1 = (const float*)d_in[4];
  const float* fW2 = (const float*)d_in[5];
  const float* fb2 = (const float*)d_in[6];
  const float* w3  = (const float*)d_in[7];
  const float* b3  = (const float*)d_in[8];
  const float* wS2 = (const float*)d_in[9];
  const float* bS2 = (const float*)d_in[10];
  float* vbuf = (float*)d_ws;  // 3 * T * B floats = ~4.7 MB scratch
  float* out  = (float*)d_out;

  // 384 single-wave workgroups: 3 s-slices x 128 batch tiles of 16
  ode_wave_kernel<<<384, 32, 0, stream>>>(x, wS1, bS1, fW1, fb1, fW2, fb2,
                                          w3, b3, wS2, bS2, vbuf);
  const int TB = T_STEPS * B_SIZE;
  softmax_kernel<<<(TB + 255) / 256, 256, 0, stream>>>(vbuf, out);
}
